// RelNet_42898133353091
// MI455X (gfx1250) — matrix operand order repaired and checked by reading, not compiled, not run on hardware
//
#include <hip/hip_runtime.h>
#include <math.h>

// ---------------------------------------------------------------------------
// Relation Network forward, fused for gfx1250 (MI455X).
//
// Layer-1 factorization: pairs@W1 = CI[i] + CJ[j] + (ques@W1q + b1), where
// CI/CJ/Q2 are tiny precomputed fp32 tables  -> removes 52 GFLOP and all
// layer-1 weight traffic. Remaining 3 g-layers (104 GFLOP over ~1.5 MB of
// weights/activations) run on v_wmma_f32_16x16x32_f16 with:
//   - each wave owning 4 m-tiles x 4 n-tiles (16 accumulators)
//   - each B fragment (global, L2-resident packed weights) reused 4x
//   - 8 LDS + 4 global loads feeding 16 independent WMMAs per k-step
// Mean over 4096 pairs via ds_add_f32 + global_atomic_add_f32.
// Tiny f-MLP + log_softmax in fp32.
// ---------------------------------------------------------------------------

typedef __attribute__((ext_vector_type(16))) _Float16 v16h;
typedef __attribute__((ext_vector_type(8)))  _Float16 v8h;
typedef __attribute__((ext_vector_type(4)))  _Float16 v4h;
typedef __attribute__((ext_vector_type(8)))  float    v8f;

#define N_IMG   64
#define OBJ     64          // H*W
#define EQ      256
#define GDIM    256
#define KP2     264         // LDS row stride in halves (conflict-free for b128)
#define TILE_M  64          // pair-rows per workgroup (fixed i, j = 0..63)

// ---------------------------- prep kernels ---------------------------------

// CI[n][o][c] = cell(n,o) @ W1[0:66]   ;  CJ[n][o][c] = cell(n,o) @ W1[66:132]
__global__ void prep_pair_tables(const float* __restrict__ img,
                                 const float* __restrict__ w1,
                                 float* __restrict__ CI, float* __restrict__ CJ) {
    int idx = blockIdx.x * blockDim.x + threadIdx.x;     // N*OBJ*GDIM
    if (idx >= N_IMG * OBJ * GDIM) return;
    int c = idx & 255, rest = idx >> 8;
    int o = rest & 63, n = rest >> 6;
    float si = 0.0f, sj = 0.0f;
    for (int d = 0; d < 64; ++d) {
        float cv = img[(n * 64 + d) * OBJ + o];          // img (N,C,H,W), o=h*8+w
        si += cv * w1[d * GDIM + c];
        sj += cv * w1[(66 + d) * GDIM + c];
    }
    float lr = (float)(o >> 3), lc = (float)(o & 7);
    si += lr * w1[64 * GDIM + c] + lc * w1[65 * GDIM + c];
    sj += lr * w1[(66 + 64) * GDIM + c] + lc * w1[(66 + 65) * GDIM + c];
    CI[idx] = si;
    CJ[idx] = sj;
}

// Q2[n][c] = b1[c] + ques[n] @ W1[132:388]
__global__ void prep_q2(const float* __restrict__ ques, const float* __restrict__ w1,
                        const float* __restrict__ b1, float* __restrict__ Q2) {
    int idx = blockIdx.x * blockDim.x + threadIdx.x;     // N*GDIM
    if (idx >= N_IMG * GDIM) return;
    int c = idx & 255, n = idx >> 8;
    float s = b1[c];
    for (int e = 0; e < EQ; ++e) s += ques[n * EQ + e] * w1[(132 + e) * GDIM + c];
    Q2[idx] = s;
}

// Pack W (256 x 256 row-major fp32) -> f16 B-fragment layout:
//   dst[ ((t*2 + h)*256 + n)*16 + tt ] = W[t*32 + h*16 + tt][n]
__global__ void prep_w(const float* __restrict__ w, _Float16* __restrict__ dst) {
    int idx = blockIdx.x * blockDim.x + threadIdx.x;     // 256*256
    if (idx >= 256 * 256) return;
    int nn = idx & 255, kp = idx >> 8;
    int t = kp >> 5, r = kp & 31, h = r >> 4, tt = r & 15;
    dst[(((t * 2 + h) * 256) + nn) * 16 + tt] = (_Float16)w[kp * 256 + nn];
}

__global__ void zero_f(float* __restrict__ p, int n) {
    int idx = blockIdx.x * blockDim.x + threadIdx.x;
    if (idx < n) p[idx] = 0.0f;
}

// ------------------------- fused g-MLP kernel -------------------------------

// One 256->256 layer. Wave `wave` owns all 4 m-tiles and n-columns
// [wave*64, wave*64+64). Each B fragment feeds 4 WMMAs (one per m-tile).
__device__ __forceinline__ void g_layer(const _Float16* __restrict__ Wp,
                                        const float* __restrict__ bias,
                                        _Float16* X, float* red, bool last,
                                        int wave, int lane) {
    const int m  = lane & 15;       // A row within tile / B,C column within tile
    const int kh = lane >> 4;       // k-half select
    const int wbase = wave * 64;    // this wave's global column base

    v8f c[4][4];
#pragma unroll
    for (int mi = 0; mi < 4; ++mi)
#pragma unroll
        for (int t = 0; t < 4; ++t)
            c[mi][t] = v8f{0.f,0.f,0.f,0.f,0.f,0.f,0.f,0.f};

#pragma unroll
    for (int ks = 0; ks < 8; ++ks) {
        // 4 A fragments (LDS, 2x ds_load_b128 each)
        v16h a[4];
#pragma unroll
        for (int mi = 0; mi < 4; ++mi) {
            const int arow = mi * 16 + m;
            const v8h lo = *(const v8h*)&X[arow * KP2 + ks * 32 + kh * 16];
            const v8h hi = *(const v8h*)&X[arow * KP2 + ks * 32 + kh * 16 + 8];
            a[mi] = __builtin_shufflevector(lo, hi,
                    0,1,2,3,4,5,6,7,8,9,10,11,12,13,14,15);
        }
        // 4 B fragments (packed global weights, one 32B read per lane each)
        v16h b[4];
#pragma unroll
        for (int t = 0; t < 4; ++t)
            b[t] = *(const v16h*)(Wp +
                    (((ks * 2 + kh) * 256) + (wbase + t * 16 + m)) * 16);
        // 16 independent WMMAs
#pragma unroll
        for (int t = 0; t < 4; ++t)
#pragma unroll
            for (int mi = 0; mi < 4; ++mi)
                c[mi][t] = __builtin_amdgcn_wmma_f32_16x16x32_f16(
                            false, a[mi], false, b[t], (short)0, c[mi][t],
                            false, false);
    }

    // C layout: VGPR r of lane -> row mi*16 + kh*8 + r, col wbase + t*16 + m
    if (!last) {
        __syncthreads();                       // readers of X are done
#pragma unroll
        for (int mi = 0; mi < 4; ++mi) {
            const int orow = mi * 16 + kh * 8;
#pragma unroll
            for (int t = 0; t < 4; ++t) {
                const int col = wbase + t * 16 + m;
                const float bv = bias[col];
#pragma unroll
                for (int r = 0; r < 8; ++r) {
                    float v = c[mi][t][r] + bv;
                    v = v > 0.0f ? v : 0.0f;
                    X[(orow + r) * KP2 + col] = (_Float16)v;
                }
            }
        }
        __syncthreads();                       // writes visible to next layer
    } else {
#pragma unroll
        for (int t = 0; t < 4; ++t) {
            const int col = wbase + t * 16 + m;
            const float bv = bias[col];
            float s = 0.0f;
#pragma unroll
            for (int mi = 0; mi < 4; ++mi)
#pragma unroll
                for (int r = 0; r < 8; ++r) {
                    float v = c[mi][t][r] + bv;
                    s += v > 0.0f ? v : 0.0f;
                }
            atomicAdd(&red[col], s);           // ds_add_f32
        }
        __syncthreads();
    }
}

__global__ __launch_bounds__(128) void rn_g_kernel(
        const float* __restrict__ CI, const float* __restrict__ CJ,
        const float* __restrict__ Q2,
        const _Float16* __restrict__ Wp2, const _Float16* __restrict__ Wp3,
        const _Float16* __restrict__ Wp4,
        const float* __restrict__ b2, const float* __restrict__ b3,
        const float* __restrict__ b4,
        float* __restrict__ context) {
    __shared__ __align__(16) _Float16 X[TILE_M * KP2];
    __shared__ float red[GDIM];

    const int tid  = threadIdx.x;
    const int lane = tid & 31, wave = tid >> 5;
    const int blk  = blockIdx.x;
    const int n  = blk >> 6;          // image
    const int ti = blk & 63;          // object i (64 consecutive pairs, j=0..63)

    // Layer 1 (factorized): X[j][c] = relu(CI[i][c] + CJ[j][c] + Q2[n][c])
    const float4* CIr = (const float4*)(CI + (n * OBJ + ti) * GDIM);
    const float4* Q2r = (const float4*)(Q2 + n * GDIM);
    for (int idx = tid; idx < TILE_M * (GDIM / 4); idx += 128) {
        const int r  = idx >> 6;                  // row (object j)
        const int c4 = idx & 63;                  // group of 4 columns
        const float4 a = CIr[c4];
        const float4 b = ((const float4*)(CJ + (n * OBJ + r) * GDIM))[c4];
        const float4 q = Q2r[c4];
        float v0 = a.x + b.x + q.x, v1 = a.y + b.y + q.y;
        float v2 = a.z + b.z + q.z, v3 = a.w + b.w + q.w;
        v4h o;
        o[0] = (_Float16)(v0 > 0.f ? v0 : 0.f);
        o[1] = (_Float16)(v1 > 0.f ? v1 : 0.f);
        o[2] = (_Float16)(v2 > 0.f ? v2 : 0.f);
        o[3] = (_Float16)(v3 > 0.f ? v3 : 0.f);
        *(v4h*)&X[r * KP2 + c4 * 4] = o;
    }
    for (int cidx = tid; cidx < GDIM; cidx += 128) red[cidx] = 0.0f;
    __syncthreads();

    g_layer(Wp2, b2, X, red, false, wave, lane);   // 256 -> 256
    g_layer(Wp3, b3, X, red, false, wave, lane);   // 256 -> 256
    g_layer(Wp4, b4, X, red, true,  wave, lane);   // 256 -> 256 + column sums

    for (int cidx = tid; cidx < GDIM; cidx += 128)
        atomicAdd(&context[n * GDIM + cidx], red[cidx]);
}

// ----------------------------- f-MLP (fp32) ---------------------------------

__global__ void f_linear(const float* __restrict__ in, const float* __restrict__ w,
                         const float* __restrict__ b, float* __restrict__ out,
                         int K, int M, float scale) {
    int idx = blockIdx.x * blockDim.x + threadIdx.x;
    if (idx >= N_IMG * M) return;
    int mm = idx % M, nn = idx / M;
    float acc = 0.0f;
    for (int k = 0; k < K; ++k) acc += in[nn * K + k] * w[k * M + mm];
    float s = b[mm] + acc * scale;
    out[idx] = s > 0.0f ? s : 0.0f;
}

__global__ void f3_logsoftmax(const float* __restrict__ y, const float* __restrict__ w,
                              const float* __restrict__ b, float* __restrict__ out) {
    int n = blockIdx.x * blockDim.x + threadIdx.x;
    if (n >= N_IMG) return;
    float s0 = b[0], s1 = b[1];
    for (int k = 0; k < 256; ++k) {
        float v = y[n * 256 + k];
        s0 += v * w[k * 2 + 0];
        s1 += v * w[k * 2 + 1];
    }
    float mx = fmaxf(s0, s1);
    float l  = mx + logf(expf(s0 - mx) + expf(s1 - mx));
    out[n * 2 + 0] = s0 - l;
    out[n * 2 + 1] = s1 - l;
}

// ------------------------------- launcher -----------------------------------

extern "C" void kernel_launch(void* const* d_in, const int* in_sizes, int n_in,
                              void* d_out, int out_size, void* d_ws, size_t ws_size,
                              hipStream_t stream) {
    const float* img  = (const float*)d_in[0];
    const float* ques = (const float*)d_in[1];
    const float* g_w1 = (const float*)d_in[2];  const float* g_b1 = (const float*)d_in[3];
    const float* g_w2 = (const float*)d_in[4];  const float* g_b2 = (const float*)d_in[5];
    const float* g_w3 = (const float*)d_in[6];  const float* g_b3 = (const float*)d_in[7];
    const float* g_w4 = (const float*)d_in[8];  const float* g_b4 = (const float*)d_in[9];
    const float* f_w1 = (const float*)d_in[10]; const float* f_b1 = (const float*)d_in[11];
    const float* f_w2 = (const float*)d_in[12]; const float* f_b2 = (const float*)d_in[13];
    const float* f_w3 = (const float*)d_in[14]; const float* f_b3 = (const float*)d_in[15];
    (void)in_sizes; (void)n_in; (void)out_size; (void)ws_size;

    char* ws = (char*)d_ws;
    float*    CI  = (float*)(ws + 0);             // 64*64*256*4 = 4,194,304 B
    float*    CJ  = (float*)(ws + 4194304);       // 4,194,304 B
    float*    Q2  = (float*)(ws + 8388608);       //    65,536 B
    _Float16* Wp2 = (_Float16*)(ws + 8454144);    //   131,072 B
    _Float16* Wp3 = (_Float16*)(ws + 8585216);
    _Float16* Wp4 = (_Float16*)(ws + 8716288);
    float*    ctx = (float*)(ws + 8847360);       //    65,536 B
    float*    y1  = (float*)(ws + 8912896);
    float*    y2  = (float*)(ws + 8978432);       // end ~9.04 MB

    prep_pair_tables<<<(N_IMG*OBJ*GDIM + 255)/256, 256, 0, stream>>>(img, g_w1, CI, CJ);
    prep_q2<<<(N_IMG*GDIM + 255)/256, 256, 0, stream>>>(ques, g_w1, g_b1, Q2);
    prep_w<<<(256*256 + 255)/256, 256, 0, stream>>>(g_w2, Wp2);
    prep_w<<<(256*256 + 255)/256, 256, 0, stream>>>(g_w3, Wp3);
    prep_w<<<(256*256 + 255)/256, 256, 0, stream>>>(g_w4, Wp4);
    zero_f<<<(N_IMG*GDIM + 255)/256, 256, 0, stream>>>(ctx, N_IMG*GDIM);

    // 262144 pair-rows / 64 per WG = 4096 workgroups, 4 waves each.
    rn_g_kernel<<<4096, 128, 0, stream>>>(CI, CJ, Q2, Wp2, Wp3, Wp4,
                                          g_b2, g_b3, g_b4, ctx);

    // f MLP: fold 1/4096 mean into first layer's input scale.
    f_linear<<<(N_IMG*256 + 255)/256, 256, 0, stream>>>(ctx, f_w1, f_b1, y1, 256, 256,
                                                        1.0f/4096.0f);
    f_linear<<<(N_IMG*256 + 255)/256, 256, 0, stream>>>(y1, f_w2, f_b2, y2, 256, 256, 1.0f);
    f3_logsoftmax<<<1, 64, 0, stream>>>(y2, f_w3, f_b3, (float*)d_out);
}
